// DFANet_70463233458429
// MI455X (gfx1250) — matrix operand order repaired and checked by reading, not compiled, not run on hardware
//
#include <hip/hip_runtime.h>
#include <hip/hip_bf16.h>
#include <math.h>

// ---------------------------------------------------------------------------
// NeuroSim-style analog PIM crossbar emulation on gfx1250 (MI455X).
//   P[i,s,k,b,f] = sum_a bits[b,s,a,i] * g[f,s,a,k]  -> per-s GEMM, WMMA f16
//   Two passes (ADC min/max is a global reduction per (i,s,k) slice):
//     pass1: WMMA tiles -> ordered-uint atomic min/max
//     pass2: WMMA recompute -> ADC quantize -> scaled accumulate -> epilogue
//   LDS-blocked: 128b x 64f workgroup tile, B staged per s, A per (s,i).
// ---------------------------------------------------------------------------

typedef _Float16 v16h __attribute__((ext_vector_type(16)));
typedef _Float16 v8h  __attribute__((ext_vector_type(8)));
typedef float    v8f  __attribute__((ext_vector_type(8)));

#define NB 1024     // batch (both layers)
#define PITCH 136   // LDS row pitch in halves (128 + 8 pad -> 2-way banks)

// ---- monotone float <-> uint order encoding for min/max atomics ----
__device__ __forceinline__ unsigned f32_ord(float f) {
  unsigned u = __float_as_uint(f);
  return (u & 0x80000000u) ? ~u : (u | 0x80000000u);
}
__device__ __forceinline__ float ord_f32(unsigned u) {
  u = (u & 0x80000000u) ? (u & 0x7fffffffu) : ~u;
  return __uint_as_float(u);
}

// ---- WMMA fragment gathers (16-bit, 16x16x32), row = per-lane row ptr ----
// A (16xK): lane<16 holds K=32kc+{0..7,16..23}; lane>=16 holds +8 / +24.
__device__ __forceinline__ v16h load_afrag(const _Float16* row, int lane,
                                           int kc) {
  int koff = kc * 32 + ((lane & 16) ? 8 : 0);
  v8h lo = *(const v8h*)(row + koff);
  v8h hi = *(const v8h*)(row + koff + 16);
  v16h r;
#pragma unroll
  for (int j = 0; j < 8; ++j) { r[j] = lo[j]; r[j + 8] = hi[j]; }
  return r;
}
// B (Kx16) stored transposed (row per output column):
// lane<16 holds K=32kc+{0..7,16..23}; lane>=16 holds +8 / +24 (same as A).
__device__ __forceinline__ v16h load_bfrag(const _Float16* row, int lane,
                                           int kc) {
  int koff = kc * 32 + ((lane & 16) ? 8 : 0);
  v8h lo = *(const v8h*)(row + koff);
  v8h hi = *(const v8h*)(row + koff + 16);
  v16h r;
#pragma unroll
  for (int j = 0; j < 8; ++j) { r[j] = lo[j]; r[j + 8] = hi[j]; }
  return r;
}

// ---- 128-thread block sum reduce ----
__device__ __forceinline__ float blk_sum128(float v, float* red, int a) {
  red[a] = v; __syncthreads();
#pragma unroll
  for (int s = 64; s > 0; s >>= 1) {
    if (a < s) red[a] += red[a + s];
    __syncthreads();
  }
  float r = red[0]; __syncthreads();
  return r;
}

// ---------------------------------------------------------------------------
// prep_x: activations -> f16 bit planes [s][i][b][128], row-sum of xq,
//         dummy-column partials D[i,s,b] = 0.3 * popcount.
// grid (NB, S), block 128
// ---------------------------------------------------------------------------
__global__ void prep_x_kernel(const float* __restrict__ x,
                              _Float16* __restrict__ bitsA,
                              float* __restrict__ Dstore,
                              float* __restrict__ xs, int N, int S) {
  __shared__ float red[128];
  int b = blockIdx.x, s = blockIdx.y, a = threadIdx.x;
  int n = s * 128 + a;
  float xv = (n < N) ? x[(size_t)b * N + n] : 0.f;
  float q = rintf(fminf(fmaxf(xv, 0.f), 1.f) * 255.f);
  int iq = (int)q;
#pragma unroll
  for (int i = 0; i < 8; ++i)
    bitsA[(((size_t)(s * 8 + i)) * NB + b) * 128 + a] =
        (_Float16)((iq >> i) & 1);
  float qs = blk_sum128(q, red, a);
  if (a == 0) atomicAdd(xs + b, qs);
#pragma unroll
  for (int i = 0; i < 8; ++i) {
    float c = blk_sum128((float)((iq >> i) & 1), red, a);
    if (a == 0) Dstore[((size_t)(i * S + s)) * NB + b] = 0.3f * c;
  }
}

// ---------------------------------------------------------------------------
// prep_g: weights+noise -> conductance slices, transposed f16 [s][k][f][128]
// grid (Fp, S), block 128
// ---------------------------------------------------------------------------
__global__ void prep_g_kernel(const float* __restrict__ w,
                              const float* __restrict__ noise,
                              _Float16* __restrict__ Gt, int Freal, int Fp,
                              int N, int Np, int S) {
  int f = blockIdx.x, s = blockIdx.y, a = threadIdx.x;
  int n = s * 128 + a;
  float g[4] = {0.f, 0.f, 0.f, 0.f};
  if (f < Freal && n < N) {
    float Xi = fminf(fmaxf(rintf((w[(size_t)f * N + n] + 1.f) * 127.5f), 0.f),
                     255.f);
    int xi = (int)Xi;
#pragma unroll
    for (int k = 0; k < 4; ++k) {
      int slc = (xi >> (2 * k)) & 3;
      float nz = noise[((size_t)f * Np + n) * 4 + k];
      g[k] = (0.9f * (float)slc + 0.3f) * (1.f + 0.05f * nz);
    }
  }
#pragma unroll
  for (int k = 0; k < 4; ++k)
    Gt[(((size_t)(s * 4 + k)) * Fp + f) * 128 + a] = (_Float16)g[k];
}

// ---------------------------------------------------------------------------
// dquant: ADC-quantize dummy column per (i,s) over b, fold into dsum[b]
// grid (8*S), block 256, 4 b's per thread.
// ---------------------------------------------------------------------------
__global__ void dquant_kernel(const float* __restrict__ Dstore,
                              float* __restrict__ dsum, int S) {
  __shared__ float rmn[256], rmx[256];
  int is = blockIdx.x, i = is / S, t = threadIdx.x;
  const float* d = Dstore + (size_t)is * NB;
  float v[4], mn = __builtin_inff(), mx = -__builtin_inff();
#pragma unroll
  for (int r = 0; r < 4; ++r) {
    v[r] = d[t + 256 * r];
    mn = fminf(mn, v[r]); mx = fmaxf(mx, v[r]);
  }
  rmn[t] = mn; rmx[t] = mx; __syncthreads();
  for (int sft = 128; sft > 0; sft >>= 1) {
    if (t < sft) {
      rmn[t] = fminf(rmn[t], rmn[t + sft]);
      rmx[t] = fmaxf(rmx[t], rmx[t + sft]);
    }
    __syncthreads();
  }
  mn = rmn[0]; mx = rmx[0];
  float st = (mx - mn) * (1.f / 32.f);
  if (!(st > 0.f)) st = 1.f;
  float rs = 1.f / st;
  float sc = (float)(1 << i) * 85.f;
#pragma unroll
  for (int r = 0; r < 4; ++r) {
    float idx = fminf(fmaxf(floorf((v[r] - mn) * rs), 0.f), 31.f);
    atomicAdd(dsum + t + 256 * r, (idx * st + mn) * sc);
  }
}

// ---- init / decode of min-max slots ----
__global__ void init_kernel(unsigned* minE, unsigned* maxE, int nmm,
                            float* dsum, float* xs) {
  int t = blockIdx.x * blockDim.x + threadIdx.x;
  if (t < nmm) { minE[t] = 0xFFFFFFFFu; maxE[t] = 0u; }
  if (t < NB) { dsum[t] = 0.f; xs[t] = 0.f; }
}
__global__ void decode_kernel(const unsigned* __restrict__ minE,
                              const unsigned* __restrict__ maxE,
                              float* __restrict__ mnstep, int nmm) {
  int t = blockIdx.x * blockDim.x + threadIdx.x;
  if (t >= nmm) return;
  float mn = ord_f32(minE[t]), mx = ord_f32(maxE[t]);
  float st = (mx - mn) * (1.f / 32.f);
  if (!(st > 0.f)) st = 1.f;
  mnstep[2 * t] = mn; mnstep[2 * t + 1] = st;
}

// ---------------------------------------------------------------------------
// LDS staging helpers (chunks of 8 halves == 16B, b128 traffic)
// ---------------------------------------------------------------------------
template <int WGF>
__device__ __forceinline__ void stage_B(_Float16* Bsh,
                                        const _Float16* __restrict__ Gt,
                                        int s, int Fp, int f0wg, int tid) {
#pragma unroll 2
  for (int c = tid; c < 4 * WGF * 16; c += 256) {
    int r = c >> 4, off = (c & 15) << 3;
    int k = r / WGF, f = r - k * WGF;
    const v8h* src =
        (const v8h*)(Gt + (((size_t)(s * 4 + k)) * Fp + f0wg + f) * 128 + off);
    *(v8h*)(Bsh + r * PITCH + off) = *src;
  }
}
template <int WGB>
__device__ __forceinline__ void stage_A(_Float16* Ash,
                                        const _Float16* __restrict__ bitsA,
                                        int s, int i, int b0wg, int tid) {
#pragma unroll 2
  for (int c = tid; c < WGB * 16; c += 256) {
    int r = c >> 4, off = (c & 15) << 3;
    const v8h* src =
        (const v8h*)(bitsA + (((size_t)(s * 8 + i)) * NB + b0wg + r) * 128 +
                     off);
    *(v8h*)(Ash + r * PITCH + off) = *src;
  }
}

// ---------------------------------------------------------------------------
// Pass 1: LDS-blocked WMMA, per-(i,s,k) min/max via ordered-uint atomics.
// block 256 = 8 waves arranged WBW x WFW; wave tile = (16*BT) x (16*FT).
// ---------------------------------------------------------------------------
template <int BT, int FT, int WBW, int WFW, int WGF>
__global__ __launch_bounds__(256) void pim_minmax_t(
    const _Float16* __restrict__ bitsA, const _Float16* __restrict__ Gt,
    unsigned* __restrict__ minE, unsigned* __restrict__ maxE, int Fp,
    int Fvalid, int S) {
  constexpr int WGB = 16 * BT * WBW;
  extern __shared__ _Float16 smem[];
  _Float16* Ash = smem;                 // WGB rows x PITCH
  _Float16* Bsh = smem + WGB * PITCH;   // 4*WGF rows x PITCH
  int tid = threadIdx.x, lane = tid & 31, wave = tid >> 5;
  int wb = wave / WFW, wf = wave - wb * WFW;
  int b0wg = blockIdx.x * WGB, f0wg = blockIdx.y * WGF;
  int mrow = lane & 15;

  for (int s = 0; s < S; ++s) {
    __syncthreads();                    // prior compute done before B re-stage
    stage_B<WGF>(Bsh, Gt, s, Fp, f0wg, tid);
    for (int i = 0; i < 8; ++i) {
      __syncthreads();                  // readers of previous A done
      stage_A<WGB>(Ash, bitsA, s, i, b0wg, tid);
      __syncthreads();                  // staging visible (covers B on i==0)

      float mnk[4], mxk[4];
#pragma unroll
      for (int k = 0; k < 4; ++k) {
        mnk[k] = __builtin_inff(); mxk[k] = -__builtin_inff();
      }
#pragma unroll
      for (int bt = 0; bt < BT; ++bt) {
        const _Float16* Ar = Ash + (wb * BT * 16 + bt * 16 + mrow) * PITCH;
        v16h af[4];
#pragma unroll
        for (int kc = 0; kc < 4; ++kc) af[kc] = load_afrag(Ar, lane, kc);
#pragma unroll
        for (int k = 0; k < 4; ++k) {
#pragma unroll
          for (int ft = 0; ft < FT; ++ft) {
            const _Float16* Br =
                Bsh + (k * WGF + wf * FT * 16 + ft * 16 + mrow) * PITCH;
            v8f c = {};
#pragma unroll
            for (int kc = 0; kc < 4; ++kc)
              c = __builtin_amdgcn_wmma_f32_16x16x32_f16(
                  false, af[kc], false, load_bfrag(Br, lane, kc), (short)0, c,
                  false, false);
            bool nv = (f0wg + wf * FT * 16 + ft * 16 + mrow) < Fvalid;
            if (nv) {
#pragma unroll
              for (int j = 0; j < 8; ++j) {
                mnk[k] = fminf(mnk[k], c[j]); mxk[k] = fmaxf(mxk[k], c[j]);
              }
            }
          }
        }
      }
#pragma unroll
      for (int k = 0; k < 4; ++k) {
        float mn = mnk[k], mx = mxk[k];
#pragma unroll
        for (int off = 16; off > 0; off >>= 1) {
          mn = fminf(mn, __shfl_xor(mn, off, 32));
          mx = fmaxf(mx, __shfl_xor(mx, off, 32));
        }
        if (lane == 0) {
          int t = (i * S + s) * 4 + k;
          atomicMin(minE + t, f32_ord(mn));
          atomicMax(maxE + t, f32_ord(mx));
        }
      }
    }
  }
}

// ---------------------------------------------------------------------------
// Pass 2: LDS-blocked WMMA recompute, ADC quantize + scaled accumulate,
// epilogue de-maps (and optional tanh).
// ---------------------------------------------------------------------------
template <int BT, int FT, int WBW, int WFW, int WGF>
__global__ __launch_bounds__(256) void pim_acc_t(
    const _Float16* __restrict__ bitsA, const _Float16* __restrict__ Gt,
    const float* __restrict__ mnstep, const float* __restrict__ dsum,
    const float* __restrict__ xs, float* __restrict__ out, int Fp, int Fvalid,
    int S, int outStride, int do_tanh) {
  constexpr int WGB = 16 * BT * WBW;
  extern __shared__ _Float16 smem[];
  _Float16* Ash = smem;
  _Float16* Bsh = smem + WGB * PITCH;
  float* mns = (float*)(Bsh + 4 * WGF * PITCH);   // 2*nmm floats
  int tid = threadIdx.x, lane = tid & 31, wave = tid >> 5;
  int wb = wave / WFW, wf = wave - wb * WFW;
  int b0wg = blockIdx.x * WGB, f0wg = blockIdx.y * WGF;
  int mrow = lane & 15;

  for (int t = tid; t < 64 * S; t += 256) mns[t] = mnstep[t];  // 2*(8*S*4)

  float acc[BT * FT * 8];
#pragma unroll
  for (int j = 0; j < BT * FT * 8; ++j) acc[j] = 0.f;

  for (int s = 0; s < S; ++s) {
    __syncthreads();
    stage_B<WGF>(Bsh, Gt, s, Fp, f0wg, tid);
    for (int i = 0; i < 8; ++i) {
      __syncthreads();
      stage_A<WGB>(Ash, bitsA, s, i, b0wg, tid);
      __syncthreads();
#pragma unroll
      for (int bt = 0; bt < BT; ++bt) {
        const _Float16* Ar = Ash + (wb * BT * 16 + bt * 16 + mrow) * PITCH;
        v16h af[4];
#pragma unroll
        for (int kc = 0; kc < 4; ++kc) af[kc] = load_afrag(Ar, lane, kc);
#pragma unroll
        for (int k = 0; k < 4; ++k) {
          int t = (i * S + s) * 4 + k;
          float mn = mns[2 * t], st = mns[2 * t + 1];
          float rs = 1.f / st;
          float scal = (float)(1 << i) * (float)(1 << (2 * k));
#pragma unroll
          for (int ft = 0; ft < FT; ++ft) {
            const _Float16* Br =
                Bsh + (k * WGF + wf * FT * 16 + ft * 16 + mrow) * PITCH;
            v8f c = {};
#pragma unroll
            for (int kc = 0; kc < 4; ++kc)
              c = __builtin_amdgcn_wmma_f32_16x16x32_f16(
                  false, af[kc], false, load_bfrag(Br, lane, kc), (short)0, c,
                  false, false);
            float* a8 = acc + ((bt * FT + ft) * 8);
#pragma unroll
            for (int j = 0; j < 8; ++j) {
              float idx = fminf(fmaxf(floorf((c[j] - mn) * rs), 0.f), 31.f);
              a8[j] += (idx * st + mn) * scal;
            }
          }
        }
      }
    }
  }
  // epilogue: de-map and store
#pragma unroll
  for (int bt = 0; bt < BT; ++bt) {
#pragma unroll
    for (int ft = 0; ft < FT; ++ft) {
      int fcol = f0wg + wf * FT * 16 + ft * 16 + mrow;
      if (fcol < Fvalid) {
        const float* a8 = acc + ((bt * FT + ft) * 8);
#pragma unroll
        for (int j = 0; j < 8; ++j) {
          int b = b0wg + wb * BT * 16 + bt * 16 + j + ((lane & 16) ? 8 : 0);
          float val = 2.f * (a8[j] - dsum[b]) * (1.f / (0.9f * 65025.f)) -
                      xs[b] * (1.f / 255.f);
          if (do_tanh) val = tanhf(val);
          out[(size_t)b * outStride + fcol] = val;
        }
      }
    }
  }
}

// ---------------------------------------------------------------------------
extern "C" void kernel_launch(void* const* d_in, const int* in_sizes, int n_in,
                              void* d_out, int out_size, void* d_ws,
                              size_t ws_size, hipStream_t stream) {
  (void)in_sizes; (void)n_in; (void)out_size; (void)ws_size;
  const float* x      = (const float*)d_in[0];
  const float* w1     = (const float*)d_in[1];
  const float* w3     = (const float*)d_in[2];
  const float* noise1 = (const float*)d_in[3];
  const float* noise3 = (const float*)d_in[4];
  float* out = (float*)d_out;

  char* ws = (char*)d_ws;
  size_t off = 0;
  auto take = [&](size_t bytes) -> void* {
    off = (off + 255) & ~(size_t)255;
    void* p = ws + off;
    off += bytes;
    return p;
  };
  _Float16* bitsA = (_Float16*)take((size_t)7 * 8 * 1024 * 128 * 2);  // 14.7MB
  _Float16* Gt    = (_Float16*)take((size_t)7 * 4 * 512 * 128 * 2);   // 3.7MB
  float* Dstore   = (float*)take((size_t)8 * 7 * 1024 * 4);
  unsigned* minE  = (unsigned*)take(256 * 4);
  unsigned* maxE  = (unsigned*)take(256 * 4);
  float* mnstep   = (float*)take(256 * 2 * 4);
  float* dsum     = (float*)take(1024 * 4);
  float* xs       = (float*)take(1024 * 4);
  float* h        = (float*)take((size_t)1024 * 512 * 4);             // 2MB

  // ---------------- layer 1: N=784 (S=7), F=512 ----------------
  {
    const int N = 784, S = 7, F = 512, Fp = 512, nmm = 8 * S * 4;
    // config: wg tile 128b x 64f, waves 2x4, wave tile 64b x 16f
    const size_t shm1 = (size_t)(128 + 4 * 64) * PITCH * 2;        // ~102 KB
    const size_t shm2 = shm1 + 2048;
    init_kernel<<<4, 256, 0, stream>>>(minE, maxE, nmm, dsum, xs);
    prep_x_kernel<<<dim3(NB, S), 128, 0, stream>>>(x, bitsA, Dstore, xs, N, S);
    prep_g_kernel<<<dim3(Fp, S), 128, 0, stream>>>(w1, noise1, Gt, F, Fp, N,
                                                   S * 128, S);
    dquant_kernel<<<8 * S, 256, 0, stream>>>(Dstore, dsum, S);
    pim_minmax_t<4, 1, 2, 4, 64><<<dim3(8, 8), 256, shm1, stream>>>(
        bitsA, Gt, minE, maxE, Fp, F, S);
    decode_kernel<<<1, 256, 0, stream>>>(minE, maxE, mnstep, nmm);
    pim_acc_t<4, 1, 2, 4, 64><<<dim3(8, 8), 256, shm2, stream>>>(
        bitsA, Gt, mnstep, dsum, xs, h, Fp, F, S, F, /*tanh=*/1);
  }
  // ---------------- layer 2: N=512 (S=4), F=10 (pad to 16) ----------------
  {
    const int N = 512, S = 4, F = 10, Fp = 16, nmm = 8 * S * 4;
    // config: wg tile 128b x 16f, waves 8x1, wave tile 16b x 16f
    const size_t shm1 = (size_t)(128 + 4 * 16) * PITCH * 2;        // ~51 KB
    const size_t shm2 = shm1 + 2048;
    init_kernel<<<4, 256, 0, stream>>>(minE, maxE, nmm, dsum, xs);
    prep_x_kernel<<<dim3(NB, S), 128, 0, stream>>>(h, bitsA, Dstore, xs, N, S);
    prep_g_kernel<<<dim3(Fp, S), 128, 0, stream>>>(w3, noise3, Gt, F, Fp, N,
                                                   S * 128, S);
    dquant_kernel<<<8 * S, 256, 0, stream>>>(Dstore, dsum, S);
    pim_minmax_t<1, 1, 8, 1, 16><<<dim3(8, 1), 256, shm1, stream>>>(
        bitsA, Gt, minE, maxE, Fp, F, S);
    decode_kernel<<<1, 256, 0, stream>>>(minE, maxE, mnstep, nmm);
    pim_acc_t<1, 1, 8, 1, 16><<<dim3(8, 1), 256, shm2, stream>>>(
        bitsA, Gt, mnstep, dsum, xs, out, Fp, F, S, F, /*tanh=*/0);
  }
}
